// LFQ_85873576116568
// MI455X (gfx1250) — matrix operand order repaired and checked by reading, not compiled, not run on hardware
//
#include <hip/hip_runtime.h>
#include <hip/hip_bf16.h>
#include <math.h>

typedef __attribute__((ext_vector_type(16))) _Float16 v16h;
typedef __attribute__((ext_vector_type(8)))  _Float16 v8h;
typedef __attribute__((ext_vector_type(8)))  float    v8f;
typedef __attribute__((ext_vector_type(4)))  float    v4f;
typedef unsigned int u32x4 __attribute__((ext_vector_type(4)));
typedef int          i32x8 __attribute__((ext_vector_type(8)));
typedef int          i32x4 __attribute__((ext_vector_type(4)));

#define NTOK   4096      // 4 * 32 * 32 tokens
#define DBITS  14
#define NCODE  16384
#define HW     1024
#define USAGE  65536
#define INV_T  100.0f    // 1 / 0.01

// ---- workspace byte offsets (total ~623 KB) ----
#define WS_ZF16   0          // 4096*32 f16 A-matrix (K padded to 32) = 262144 B
#define WS_OFFS   262144     // 4096 f32 per-token softmax offsets
#define WS_AVG    278528     // 16384 f32 avg_probs column sums
#define WS_IDX    344064     // 4096 i32 indices
#define WS_HIST   360448     // 65536 i32 histogram
#define WS_SCAL   622592     // f32 commit_sum, f32 sent_sum, u32 nz_count

// ---- output float offsets ----
#define OUT_Q      0
#define OUT_COMMIT 57344
#define OUT_AUX    57345
#define OUT_USAGE  57346
#define OUT_IDX    57347

#if __has_builtin(__builtin_amdgcn_tensor_load_to_lds)
#define LFQ_HAVE_TDM 1
#endif

// ------------------------------------------------------------------
// Stage one 1KB A-tile (16 tokens x 32 f16) from global into LDS.
// Preferred: Tensor Data Mover (one instruction, TENSORcnt-tracked).
// Fallback: async global->LDS B128 copies (ASYNCcnt-tracked).
// Called by wave 0 only.
// ------------------------------------------------------------------
static __device__ __forceinline__ void lfq_stage_tile(unsigned lds_off,
                                                      const _Float16* gsrc,
                                                      int lane) {
#if defined(LFQ_HAVE_TDM)
  (void)lane;
  unsigned long long ga = (unsigned long long)(uintptr_t)gsrc;
  u32x4 g0;
  g0[0] = 1u;                                   // count=1 (valid descriptor)
  g0[1] = lds_off;                              // lds_addr (bytes)
  g0[2] = (unsigned)(ga & 0xFFFFFFFFull);       // global_addr[31:0]
  g0[3] = (unsigned)((ga >> 32) & 0x01FFFFFFull) | (2u << 30); // [56:32]|type=2
  i32x8 g1 = {0, 0, 0, 0, 0, 0, 0, 0};
  g1[0] = 3 << 16;                // data_size=3 (8 bytes)
  g1[1] = 0;                      // no atomic barrier
  g1[2] = 0x0010 | (1 << 16);     // tensor_dim0=2^20 (no OOB), tensor_dim1=1
  g1[3] = 128 << 16;              // tile_dim0 = 128 * 8B = 1 KB
  g1[5] = 128;                    // tensor_dim0_stride (unused for 1 row)
  i32x4 z4 = {0, 0, 0, 0};
#if defined(__clang_major__) && (__clang_major__ >= 23)
  i32x8 z8 = {0, 0, 0, 0, 0, 0, 0, 0};
  __builtin_amdgcn_tensor_load_to_lds(g0, g1, z4, z4, z8, 0);
#else
  __builtin_amdgcn_tensor_load_to_lds(g0, g1, z4, z4, 0);
#endif
#else
  // 32 lanes x 2 x 16B = 1 KB; inst offset applies to both LDS and global.
  unsigned l0 = lds_off + (unsigned)lane * 32u;
  unsigned long long ga =
      (unsigned long long)(uintptr_t)gsrc + (unsigned long long)lane * 32u;
  asm volatile("global_load_async_to_lds_b128 %0, %1, off"
               :: "v"(l0), "v"(ga) : "memory");
  asm volatile("global_load_async_to_lds_b128 %0, %1, off offset:16"
               :: "v"(l0), "v"(ga) : "memory");
#endif
}

static __device__ __forceinline__ void lfq_stage_wait() {
#if defined(LFQ_HAVE_TDM)
#if __has_builtin(__builtin_amdgcn_s_wait_tensorcnt)
  __builtin_amdgcn_s_wait_tensorcnt(0);
#else
  asm volatile("s_wait_tensorcnt 0x0" ::: "memory");
#endif
#else
  asm volatile("s_wait_asynccnt 0x0" ::: "memory");
#endif
}

// ------------------------------------------------------------------
// Kernel 0: zero histogram + scalar accumulators (must run each call)
// ------------------------------------------------------------------
__global__ void lfq_init(int* __restrict__ hist, float* __restrict__ scal,
                         unsigned* __restrict__ nz) {
  int t = blockIdx.x * 256 + threadIdx.x;
  if (t < USAGE) hist[t] = 0;
  if (t == 0) { scal[0] = 0.0f; scal[1] = 0.0f; *nz = 0u; }
}

// ------------------------------------------------------------------
// Kernel 1: per-token pass. sign quantize, idx, commit loss,
// analytic softmax offset o_t = scaled_max + log D (separable over bits),
// analytic per-sample entropy, packs A = (2/T)*z into f16 (K padded to 32).
// ------------------------------------------------------------------
__global__ void lfq_token(const float* __restrict__ z, float* __restrict__ out,
                          _Float16* __restrict__ zf16, float* __restrict__ offs,
                          int* __restrict__ idxi, float* __restrict__ scal) {
  const int t = blockIdx.x * 256 + threadIdx.x;   // 0..4095
  const int b = t >> 10;
  const int pix = t & 1023;
  const float* zb = z + (size_t)b * DBITS * HW + pix;
  float* qb = out + OUT_Q + (size_t)b * DBITS * HW + pix;

  float commit = 0.0f, absum = 0.0f, logD = 0.0f, sent = 0.0f;
  int idx = 0;
  #pragma unroll
  for (int k = 0; k < DBITS; ++k) {
    float zv = zb[k * HW];
    float q  = zv > 0.0f ? 1.0f : -1.0f;
    qb[k * HW] = q;                          // straight-through forward value
    if (zv > 0.0f) idx |= (1 << k);
    float d = zv - q; commit += d * d;
    float az = fabsf(zv); absum += az;
    float c = 4.0f * INV_T * az;             // logit gap of flipping bit k
    float e = __expf(-c);
    logD += __logf(1.0f + e);                // log softmax denominator
    sent += c * e / (1.0f + e);              // -S/D term of sample entropy
    zf16[(size_t)t * 32 + k] = (_Float16)(2.0f * INV_T * zv);
  }
  #pragma unroll
  for (int k = DBITS; k < 32; ++k) zf16[(size_t)t * 32 + k] = (_Float16)0.0f;

  offs[t] = 2.0f * INV_T * absum + logD;     // scaled rowmax + log denom
  idxi[t] = idx;
  out[OUT_IDX + t] = (float)idx;
  atomicAdd(&scal[0], commit);
  atomicAdd(&scal[1], logD + sent);          // exact -sum p*log p for token t
}

// ------------------------------------------------------------------
// Kernel 2: avg_probs via WMMA with LDS-staged, double-buffered A tiles.
// 8 waves/block, each wave owns 16 codebook columns (128 cols/block).
// Wave 0 TDM/async-prefetches A chunk m+1 while all waves compute chunk m:
//   u = A(16x32 f16) x B(32x16 f16) via v_wmma_f32_16x16x32_f16,
//   p = exp(u - o_t), column-accumulate over all 4096 tokens.
// B (+/-1 code bits) generated in-register once; offsets resident in LDS.
// ------------------------------------------------------------------
__global__ void __launch_bounds__(256)
lfq_wmma_avgprobs(const _Float16* __restrict__ zf16,
                  const float* __restrict__ offs,
                  float* __restrict__ avgsum) {
  __shared__ float    s_offs[NTOK];        // 16 KB: all per-token offsets
  __shared__ _Float16 s_a[2][16 * 32];     // 2 KB: double-buffered A tiles

  const int tid  = threadIdx.x;
  const int lane = tid & 31;
  const int wave = tid >> 5;
  const int nbase = blockIdx.x * 128 + wave * 16;
  const bool hi = (lane & 16) != 0;

  // Kick off stage of chunk 0 ASAP (wave 0), overlap with offset preload.
  const unsigned lds_a0 = (unsigned)(uintptr_t)&s_a[0][0];
  const unsigned lds_a1 = (unsigned)(uintptr_t)&s_a[1][0];
  if (wave == 0) lfq_stage_tile(lds_a0, zf16, lane);

  for (int j = tid; j < NTOK; j += 256) s_offs[j] = offs[j];

  // B-matrix 32x16 (KxN): lanes 0-15 hold K{0-7,16-23} for col=lane,
  // lanes 16-31 hold K{8-15,24-31} for col=lane-16. Codebook value = bit ? +1 : -1.
  v16h bmat;
  const int col = nbase + (lane & 15);
  const int kb0 = hi ? 8 : 0;
  #pragma unroll
  for (int i = 0; i < 16; ++i) {
    int k = kb0 + (i < 8 ? i : i + 8);
    float s = (k < DBITS) ? (((col >> k) & 1) ? 1.0f : -1.0f) : 0.0f;
    bmat[i] = (_Float16)s;
  }

  float acc = 0.0f;
  const int chunkA = hi ? 1 : 0;  // A K-groups: lanes<16 K{0-7,16-23}, hi K{8-15,24-31}
  for (int mc = 0; mc < NTOK / 16; ++mc) {
    if (wave == 0) lfq_stage_wait();        // staged chunk mc has landed
    __syncthreads();                        // ...and everyone finished mc-1 reads
    if (wave == 0 && mc + 1 < NTOK / 16)    // prefetch next chunk into other buf
      lfq_stage_tile((mc & 1) ? lds_a0 : lds_a1,
                     zf16 + (size_t)(mc + 1) * 16 * 32, lane);

    const int mbase = mc * 16;
    const _Float16* arow = &s_a[mc & 1][(lane & 15) * 32];
    v8h a0 = ((const v8h*)arow)[chunkA];    // ds_load_b128
    v8h a1 = ((const v8h*)arow)[chunkA + 2];
    v16h amat;
    #pragma unroll
    for (int i = 0; i < 8; ++i) { amat[i] = a0[i]; amat[8 + i] = a1[i]; }

    const v4f* op = (const v4f*)&s_offs[mbase + (hi ? 8 : 0)];
    v4f o0 = op[0];
    v4f o1 = op[1];

    v8f c = {};
    c = __builtin_amdgcn_wmma_f32_16x16x32_f16(false, amat, false, bmat,
                                               (short)0, c, false, false);
    float oa[8] = {o0[0], o0[1], o0[2], o0[3], o1[0], o1[1], o1[2], o1[3]};
    #pragma unroll
    for (int v = 0; v < 8; ++v) acc += __expf(c[v] - oa[v]);
  }

  // combine the two half-wave partial sums for each column (wave32)
  float tot = acc + __shfl_down(acc, 16, 32);
  if (lane < 16) avgsum[nbase + (lane & 15)] = tot;
}

// ------------------------------------------------------------------
// Kernel 3: entropy of avg_probs + final loss scalars
// ------------------------------------------------------------------
__global__ void lfq_finalize_losses(const float* __restrict__ avgsum,
                                    const float* __restrict__ scal,
                                    float* __restrict__ out) {
  __shared__ float red[256];
  float s = 0.0f;
  for (int j = threadIdx.x; j < NCODE; j += 256) {
    float ap = avgsum[j] * (1.0f / (float)NTOK);
    s += ap * __logf(ap + 1e-5f);
  }
  red[threadIdx.x] = s;
  __syncthreads();
  for (int st = 128; st > 0; st >>= 1) {
    if (threadIdx.x < st) red[threadIdx.x] += red[threadIdx.x + st];
    __syncthreads();
  }
  if (threadIdx.x == 0) {
    float avg_entropy = -red[0];
    float sample_entropy = scal[1] / (float)NTOK;
    out[OUT_AUX] = 0.1f * (sample_entropy - avg_entropy);
    out[OUT_COMMIT] = 0.25f * scal[0] / (float)(NTOK * DBITS);
  }
}

// ------------------------------------------------------------------
// Kernels 4: ring-buffer usage histogram
// ------------------------------------------------------------------
__global__ void lfq_hist_fill(const int* __restrict__ used,
                              const int* __restrict__ idxi,
                              int* __restrict__ hist) {
  int t = blockIdx.x * 256 + threadIdx.x;     // 0..65535
  int v = (t < USAGE - NTOK) ? used[NTOK + t] : idxi[t - (USAGE - NTOK)];
  atomicAdd(&hist[((unsigned)v) & (USAGE - 1)], 1);
}

__global__ void lfq_hist_count(const int* __restrict__ hist,
                               unsigned* __restrict__ nz) {
  __shared__ unsigned red[256];
  unsigned c = 0;
  for (int j = blockIdx.x * 256 + threadIdx.x; j < USAGE; j += 64 * 256)
    c += (hist[j] > 0) ? 1u : 0u;
  red[threadIdx.x] = c;
  __syncthreads();
  for (int st = 128; st > 0; st >>= 1) {
    if (threadIdx.x < st) red[threadIdx.x] += red[threadIdx.x + st];
    __syncthreads();
  }
  if (threadIdx.x == 0) atomicAdd(nz, red[0]);
}

__global__ void lfq_usage_write(const unsigned* __restrict__ nz,
                                float* __restrict__ out) {
  out[OUT_USAGE] = (float)(*nz) / (float)NCODE;
}

// ------------------------------------------------------------------
extern "C" void kernel_launch(void* const* d_in, const int* in_sizes, int n_in,
                              void* d_out, int out_size, void* d_ws, size_t ws_size,
                              hipStream_t stream) {
  (void)in_sizes; (void)n_in; (void)out_size; (void)ws_size;
  const float* z    = (const float*)d_in[0];
  // d_in[1] (codebook) is not loaded: its rows are exactly the +/-1 bit
  // patterns of the row index, regenerated in-register in the WMMA kernel.
  const int*   used = (const int*)d_in[2];
  float* out = (float*)d_out;

  char* w = (char*)d_ws;
  _Float16* zf16 = (_Float16*)(w + WS_ZF16);
  float*    offs = (float*)   (w + WS_OFFS);
  float*    avgs = (float*)   (w + WS_AVG);
  int*      idxi = (int*)     (w + WS_IDX);
  int*      hist = (int*)     (w + WS_HIST);
  float*    scal = (float*)   (w + WS_SCAL);
  unsigned* nz   = (unsigned*)(w + WS_SCAL + 8);

  lfq_init            <<<256, 256, 0, stream>>>(hist, scal, nz);
  lfq_token           <<< 16, 256, 0, stream>>>(z, out, zf16, offs, idxi, scal);
  lfq_wmma_avgprobs   <<<128, 256, 0, stream>>>(zf16, offs, avgs);
  lfq_finalize_losses <<<  1, 256, 0, stream>>>(avgs, scal, out);
  lfq_hist_fill       <<<256, 256, 0, stream>>>(used, idxi, hist);
  lfq_hist_count      <<< 64, 256, 0, stream>>>(hist, nz);
  lfq_usage_write     <<<  1,   1, 0, stream>>>(nz, out);
}
